// TransformerBlock_61641370632932
// MI455X (gfx1250) — matrix operand order repaired and checked by reading, not compiled, not run on hardware
//
#include <hip/hip_runtime.h>
#include <hip/hip_bf16.h>
#include <math.h>

// ---------------- problem constants ----------------
#define D_MODEL 768
#define NHEAD   12
#define HEAD_DIM 64
#define DFF     3072
#define BATCH   2
#define SEQ     2048
#define MROWS   (BATCH * SEQ)   // 4096

typedef __attribute__((ext_vector_type(16))) _Float16 v16h;
typedef __attribute__((ext_vector_type(8)))  _Float16 v8h;
typedef __attribute__((ext_vector_type(8)))  float    v8f;

static __device__ __forceinline__ v16h cat8(v8h lo, v8h hi) {
  return __builtin_shufflevector(lo, hi, 0,1,2,3,4,5,6,7,8,9,10,11,12,13,14,15);
}

static __device__ __forceinline__ v8f wmma_f16(v16h a, v16h b, v8f c) {
  // D = A(16x32 f16) * B(32x16 f16) + C(16x16 f32)
  return __builtin_amdgcn_wmma_f32_16x16x32_f16(false, a, false, b, (short)0, c, false, false);
}

// ---------------- fp32 -> f16 weight conversion ----------------
__global__ void cvt_f32_f16_kernel(const float* __restrict__ in,
                                   _Float16* __restrict__ out, int n) {
  int i = blockIdx.x * 256 + threadIdx.x;
  if (i < n) out[i] = (_Float16)in[i];
}

// ---------------- LayerNorm (row of 768) -> f16 ----------------
__global__ __launch_bounds__(256)
void ln_f16_kernel(const float* __restrict__ x, const float* __restrict__ g,
                   const float* __restrict__ be, _Float16* __restrict__ out) {
  const int row = blockIdx.x;
  const float* xr = x + (size_t)row * D_MODEL;
  float v[3];
  float s = 0.f, sq = 0.f;
#pragma unroll
  for (int j = 0; j < 3; ++j) {
    v[j] = xr[threadIdx.x + j * 256];
    s += v[j];
    sq += v[j] * v[j];
  }
#pragma unroll
  for (int off = 16; off; off >>= 1) {
    s  += __shfl_xor(s, off, 32);
    sq += __shfl_xor(sq, off, 32);
  }
  __shared__ float ss[8], ssq[8];
  const int w = threadIdx.x >> 5;
  if ((threadIdx.x & 31) == 0) { ss[w] = s; ssq[w] = sq; }
  __syncthreads();
  s = 0.f; sq = 0.f;
#pragma unroll
  for (int j = 0; j < 8; ++j) { s += ss[j]; sq += ssq[j]; }
  const float mu  = s * (1.0f / D_MODEL);
  const float var = sq * (1.0f / D_MODEL) - mu * mu;
  const float r   = rsqrtf(var + 1e-5f);
#pragma unroll
  for (int j = 0; j < 3; ++j) {
    const int c = threadIdx.x + j * 256;
    out[(size_t)row * D_MODEL + c] = (_Float16)((v[j] - mu) * r * g[c] + be[c]);
  }
}

// ---------------- generic WMMA GEMM: C[M,N] = A[M,K] * W[N,K]^T ----------------
// Block: 256 threads = 8 waves as 2x4; block tile 64x256; wave tile 32x64.
// epi: 0 = (c+bias) -> f16 out
//      1 = gelu(c+bias) -> f16 out
//      2 = (c+bias+resid) -> f32 out
__global__ __launch_bounds__(256)
void gemm_wmma_kernel(const _Float16* __restrict__ A, const _Float16* __restrict__ W,
                      const float* __restrict__ bias, const float* __restrict__ resid,
                      float* __restrict__ outF, _Float16* __restrict__ outH,
                      int M, int N, int K, int epi) {
  const int lane = threadIdx.x & 31;
  const int w    = threadIdx.x >> 5;
  const int mw   = w >> 2;          // 0..1
  const int nw   = w & 3;           // 0..3
  const int row0 = blockIdx.y * 64 + mw * 32;
  const int col0 = blockIdx.x * 256 + nw * 64;
  const int lm   = lane & 15;
  const int sel  = lane >> 4;

  v8f acc[2][4] = {};

  const _Float16* Abase = A + (size_t)(row0 + lm) * K + sel * 16;
  const _Float16* Wbase = W + (size_t)(col0 + lm) * K + sel * 16;

  for (int k0 = 0; k0 < K; k0 += 32) {
    if (k0 + 128 <= K) {
      // keep upcoming tiles warm near the WGP (locality 3 -> RT near-cache policy)
      __builtin_prefetch(Abase + k0 + 128, 0, 3);
      __builtin_prefetch(Wbase + k0 + 128, 0, 3);
    }
    v16h a[2], b[4];
#pragma unroll
    for (int mi = 0; mi < 2; ++mi) {
      const _Float16* p = Abase + (size_t)mi * 16 * K + k0;
      a[mi] = cat8(*(const v8h*)p, *(const v8h*)(p + 8));
    }
#pragma unroll
    for (int ni = 0; ni < 4; ++ni) {
      const _Float16* p = Wbase + (size_t)ni * 16 * K + k0;
      b[ni] = cat8(*(const v8h*)p, *(const v8h*)(p + 8));
    }
#pragma unroll
    for (int mi = 0; mi < 2; ++mi)
#pragma unroll
      for (int ni = 0; ni < 4; ++ni)
        acc[mi][ni] = wmma_f16(a[mi], b[ni], acc[mi][ni]);
  }

#pragma unroll
  for (int mi = 0; mi < 2; ++mi) {
#pragma unroll
    for (int ni = 0; ni < 4; ++ni) {
#pragma unroll
      for (int i = 0; i < 8; ++i) {
        const int r = row0 + mi * 16 + i + sel * 8;
        const int c = col0 + ni * 16 + lm;
        float vv = acc[mi][ni][i] + bias[c];
        const size_t idx = (size_t)r * N + c;
        if (epi == 1) {
          vv = 0.5f * vv * (1.0f + erff(vv * 0.70710678118654752f));
          outH[idx] = (_Float16)vv;
        } else if (epi == 2) {
          outF[idx] = vv + resid[idx];
        } else {
          outH[idx] = (_Float16)vv;
        }
      }
    }
  }
}

// ---------------- V transpose: [B*T, 768] (head-sliced) -> Vt[B,H,64,T] ----------------
__global__ void v_transpose_kernel(const _Float16* __restrict__ Vh,
                                   _Float16* __restrict__ Vt) {
  const int idx = blockIdx.x * 256 + threadIdx.x;   // over B*H*64*SEQ
  const int t    = idx & (SEQ - 1);
  const int rest = idx >> 11;
  const int d    = rest & (HEAD_DIM - 1);
  const int bh   = rest >> 6;
  const int h = bh % NHEAD;
  const int b = bh / NHEAD;
  Vt[idx] = Vh[((size_t)(b * SEQ + t)) * D_MODEL + h * HEAD_DIM + d];
}

// ---------------- flash attention w/ ALiBi: one wave per 16-query tile ----------------
__global__ __launch_bounds__(128)
void attn_kernel(const _Float16* __restrict__ Qh, const _Float16* __restrict__ Kh,
                 const _Float16* __restrict__ Vt, _Float16* __restrict__ Ctx) {
  __shared__ _Float16 pb[4][16][40];     // per-wave 16x32 P tile (padded rows)
  const int lane = threadIdx.x & 31;
  const int w    = threadIdx.x >> 5;
  const int lm   = lane & 15;
  const int sel  = lane >> 4;
  const int bh = blockIdx.y;
  const int b  = bh / NHEAD;
  const int h  = bh % NHEAD;
  const int q0 = (blockIdx.x * 4 + w) * 16;
  const float slope = (h < 8) ? exp2f(-(float)(h + 1))
                              : exp2f(-0.5f * (2.0f * (float)(h - 8) + 1.0f));

  const size_t rowbase = (size_t)(b * SEQ) * D_MODEL + h * HEAD_DIM;

  // Q A-fragments for k-dim 0..31 and 32..63 (resident for whole pass)
  v16h qa[2];
  {
    const _Float16* p = Qh + rowbase + (size_t)(q0 + lm) * D_MODEL + sel * 16;
    qa[0] = cat8(*(const v8h*)p,        *(const v8h*)(p + 8));
    qa[1] = cat8(*(const v8h*)(p + 32), *(const v8h*)(p + 40));
  }

  v8f o[4] = {};
  float mrow[8], lrow[8];
#pragma unroll
  for (int i = 0; i < 8; ++i) { mrow[i] = -1e30f; lrow[i] = 0.f; }

  for (int kt0 = 0; kt0 < SEQ; kt0 += 32) {
    // ---- scores S = Q K^T for a 16x32 block ----
    v8f s0 = {}, s1 = {};
#pragma unroll
    for (int dstep = 0; dstep < 2; ++dstep) {
      const _Float16* p0 = Kh + rowbase + (size_t)(kt0 + lm) * D_MODEL + dstep * 32 + sel * 16;
      const _Float16* p1 = p0 + (size_t)16 * D_MODEL;
      v16h kb0 = cat8(*(const v8h*)p0, *(const v8h*)(p0 + 8));
      v16h kb1 = cat8(*(const v8h*)p1, *(const v8h*)(p1 + 8));
      s0 = wmma_f16(qa[dstep], kb0, s0);
      s1 = wmma_f16(qa[dstep], kb1, s1);
    }
    // ---- scale + alibi + online softmax (rows live in 16-lane halves) ----
#pragma unroll
    for (int i = 0; i < 8; ++i) {
      const float qabs = (float)(q0 + i + sel * 8);
      float e0 = s0[i] * 0.125f - slope * fabsf(qabs - (float)(kt0 + lm));
      float e1 = s1[i] * 0.125f - slope * fabsf(qabs - (float)(kt0 + 16 + lm));
      float tmax = fmaxf(e0, e1);
#pragma unroll
      for (int off = 8; off; off >>= 1) tmax = fmaxf(tmax, __shfl_xor(tmax, off, 32));
      const float mnew = fmaxf(mrow[i], tmax);
      const float corr = __expf(mrow[i] - mnew);
      const float p0v = __expf(e0 - mnew);
      const float p1v = __expf(e1 - mnew);
      float rs = p0v + p1v;
#pragma unroll
      for (int off = 8; off; off >>= 1) rs += __shfl_xor(rs, off, 32);
      lrow[i] = lrow[i] * corr + rs;
      mrow[i] = mnew;
#pragma unroll
      for (int t = 0; t < 4; ++t) o[t][i] *= corr;
      // stash P in C-layout into LDS for layout swap
      pb[w][i + sel * 8][lm]      = (_Float16)p0v;
      pb[w][i + sel * 8][16 + lm] = (_Float16)p1v;
    }
    asm volatile("s_wait_dscnt 0x0" ::: "memory");
    // reload P as an A-fragment (16x32, k = key dim)
    v16h pa;
    {
      const _Float16* p = &pb[w][lm][sel * 16];
      pa = cat8(*(const v8h*)p, *(const v8h*)(p + 8));
    }
    // ---- O += P * V  (V^T gives contiguous B-fragments) ----
#pragma unroll
    for (int t = 0; t < 4; ++t) {
      const _Float16* p = Vt + ((size_t)bh * HEAD_DIM + t * 16 + lm) * SEQ + kt0 + sel * 16;
      v16h vb = cat8(*(const v8h*)p, *(const v8h*)(p + 8));
      o[t] = wmma_f16(pa, vb, o[t]);
    }
  }

  // ---- finalize: divide by softmax denom, write ctx back in [B*T, 768] f16 ----
#pragma unroll
  for (int t = 0; t < 4; ++t) {
#pragma unroll
    for (int i = 0; i < 8; ++i) {
      const float val = o[t][i] / lrow[i];
      const size_t r = (size_t)(b * SEQ + q0 + i + sel * 8);
      Ctx[r * D_MODEL + h * HEAD_DIM + t * 16 + lm] = (_Float16)val;
    }
  }
}

// ---------------- host-side orchestration ----------------
extern "C" void kernel_launch(void* const* d_in, const int* in_sizes, int n_in,
                              void* d_out, int out_size, void* d_ws, size_t ws_size,
                              hipStream_t stream) {
  const float* x   = (const float*)d_in[0];
  const float* Wq  = (const float*)d_in[1];
  const float* bq  = (const float*)d_in[2];
  const float* Wk  = (const float*)d_in[3];
  const float* bk  = (const float*)d_in[4];
  const float* Wv  = (const float*)d_in[5];
  const float* bv  = (const float*)d_in[6];
  const float* Wo  = (const float*)d_in[7];
  const float* bo  = (const float*)d_in[8];
  const float* W1  = (const float*)d_in[9];
  const float* b1  = (const float*)d_in[10];
  const float* W2  = (const float*)d_in[11];
  const float* b2  = (const float*)d_in[12];
  const float* g1  = (const float*)d_in[13];
  const float* be1 = (const float*)d_in[14];
  const float* g2  = (const float*)d_in[15];
  const float* be2 = (const float*)d_in[16];
  float* out = (float*)d_out;

  // workspace carve-up (256B aligned)
  size_t off = 0;
  char* base = (char*)d_ws;
  auto alloc = [&](size_t bytes) -> void* {
    void* p = base + off;
    off += (bytes + 255) & ~(size_t)255;
    return p;
  };
  const size_t DD  = (size_t)D_MODEL * D_MODEL;          // 768*768
  const size_t FD  = (size_t)DFF * D_MODEL;              // 3072*768
  const size_t MD  = (size_t)MROWS * D_MODEL;            // 4096*768
  const size_t MF  = (size_t)MROWS * DFF;                // 4096*3072

  _Float16* Wq_h = (_Float16*)alloc(DD * 2);
  _Float16* Wk_h = (_Float16*)alloc(DD * 2);
  _Float16* Wv_h = (_Float16*)alloc(DD * 2);
  _Float16* Wo_h = (_Float16*)alloc(DD * 2);
  _Float16* W1_h = (_Float16*)alloc(FD * 2);
  _Float16* W2_h = (_Float16*)alloc(FD * 2);
  _Float16* h1   = (_Float16*)alloc(MD * 2);
  _Float16* Qh   = (_Float16*)alloc(MD * 2);
  _Float16* Kh   = (_Float16*)alloc(MD * 2);
  _Float16* Vh   = (_Float16*)alloc(MD * 2);
  _Float16* Vt   = (_Float16*)alloc(MD * 2);
  _Float16* Ctx  = (_Float16*)alloc(MD * 2);
  float*    x1   = (float*)   alloc(MD * 4);
  _Float16* h2   = (_Float16*)alloc(MD * 2);
  _Float16* ff1  = (_Float16*)alloc(MF * 2);
  (void)ws_size; (void)n_in; (void)in_sizes; (void)out_size;

  // 1) weights -> f16
  auto cvt = [&](const float* src, _Float16* dst, size_t n) {
    cvt_f32_f16_kernel<<<(unsigned)((n + 255) / 256), 256, 0, stream>>>(src, dst, (int)n);
  };
  cvt(Wq, Wq_h, DD); cvt(Wk, Wk_h, DD); cvt(Wv, Wv_h, DD); cvt(Wo, Wo_h, DD);
  cvt(W1, W1_h, FD); cvt(W2, W2_h, FD);

  // 2) h1 = LN(x; g1, be1) in f16
  ln_f16_kernel<<<MROWS, 256, 0, stream>>>(x, g1, be1, h1);

  // 3) QKV projections (M=4096, N=768, K=768)
  {
    dim3 g(D_MODEL / 256, MROWS / 64);
    gemm_wmma_kernel<<<g, 256, 0, stream>>>(h1, Wq_h, bq, nullptr, nullptr, Qh,
                                            MROWS, D_MODEL, D_MODEL, 0);
    gemm_wmma_kernel<<<g, 256, 0, stream>>>(h1, Wk_h, bk, nullptr, nullptr, Kh,
                                            MROWS, D_MODEL, D_MODEL, 0);
    gemm_wmma_kernel<<<g, 256, 0, stream>>>(h1, Wv_h, bv, nullptr, nullptr, Vh,
                                            MROWS, D_MODEL, D_MODEL, 0);
  }

  // 4) V transpose to [B,H,64,T]
  v_transpose_kernel<<<(unsigned)(MD / 256), 256, 0, stream>>>(Vh, Vt);

  // 5) attention -> Ctx f16 [B*T, 768]
  {
    dim3 g(SEQ / 16 / 4, BATCH * NHEAD);
    attn_kernel<<<g, 128, 0, stream>>>(Qh, Kh, Vt, Ctx);
  }

  // 6) x1 = x + Ctx @ Wo^T + bo  (fp32)
  {
    dim3 g(D_MODEL / 256, MROWS / 64);
    gemm_wmma_kernel<<<g, 256, 0, stream>>>(Ctx, Wo_h, bo, x, x1, nullptr,
                                            MROWS, D_MODEL, D_MODEL, 2);
  }

  // 7) h2 = LN(x1; g2, be2) in f16
  ln_f16_kernel<<<MROWS, 256, 0, stream>>>(x1, g2, be2, h2);

  // 8) ff1 = gelu(h2 @ W1^T + b1) f16  (M=4096, N=3072, K=768)
  {
    dim3 g(DFF / 256, MROWS / 64);
    gemm_wmma_kernel<<<g, 256, 0, stream>>>(h2, W1_h, b1, nullptr, nullptr, ff1,
                                            MROWS, DFF, D_MODEL, 1);
  }

  // 9) out = x1 + ff1 @ W2^T + b2  (M=4096, N=768, K=3072)
  {
    dim3 g(D_MODEL / 256, MROWS / 64);
    gemm_wmma_kernel<<<g, 256, 0, stream>>>(ff1, W2_h, b2, x1, out, nullptr,
                                            MROWS, D_MODEL, DFF, 2);
  }
}
